// GraphTransformer_32925219291880
// MI455X (gfx1250) — compile-verified, matrix operand order checked by reading, not run on hardware
//
#include <hip/hip_runtime.h>
#include <hip/hip_bf16.h>
#include <math.h>

typedef __attribute__((ext_vector_type(8)))  _Float16 v8h;
typedef __attribute__((ext_vector_type(16))) _Float16 v16h;
typedef __attribute__((ext_vector_type(8)))  float    v8f;
typedef __attribute__((ext_vector_type(4)))  unsigned int u32x4;
typedef __attribute__((ext_vector_type(8)))  int i32x8;
typedef __attribute__((ext_vector_type(4)))  int i32x4;

#define N_NODES 10000
#define N_EDGES 160000
#define DIM_IN  256
#define DD      64
#define NHEADS  8
#define HCW     512   // heads * channels for stacked layers

#if defined(__has_builtin)
# if __has_builtin(__builtin_amdgcn_tensor_load_to_lds) && \
     __has_builtin(__builtin_amdgcn_s_wait_tensorcnt)
#  define HAVE_TDM 1
# endif
#endif
#ifndef HAVE_TDM
# define HAVE_TDM 0
#endif

// ---------------------------------------------------------------------------
// Weight pre-pack: W[K,M] f32 -> f16 tiles in WMMA B-fragment order.
// Tile (kt, ct) covers K rows [kt*32, kt*32+32), cols [ct*16, ct*16+16).
// Within a tile (512 halves): lane L owns 16 contiguous halves; element e maps
// to k_local = (e<8 ? kbase+e : 16+kbase+e-8), kbase = (L>=16)*8, col = L%16.
// A 4-tile column group (64 cols) for one k-step is 4KB *contiguous*.
// ---------------------------------------------------------------------------
__global__ void pack_w_kernel(const float* __restrict__ W, _Float16* __restrict__ P,
                              int K, int M) {
  size_t i = (size_t)blockIdx.x * blockDim.x + threadIdx.x;
  if (i >= (size_t)K * M) return;
  const int    e    = (int)(i & 15);
  const int    L    = (int)((i >> 4) & 31);
  const size_t tile = i >> 9;             // /512
  const int    mt   = M >> 4;
  const int    ct   = (int)(tile % mt);
  const int    kt   = (int)(tile / mt);
  const int    kbase  = (L >> 4) * 8;
  const int    klocal = (e < 8) ? (kbase + e) : (16 + kbase + (e - 8));
  const int    k   = kt * 32 + klocal;
  const int    col = ct * 16 + (L & 15);
  P[i] = (_Float16)W[(size_t)k * M + col];
}

__global__ void cvt_f16_kernel(const float* __restrict__ in, _Float16* __restrict__ out,
                               size_t n) {
  size_t i = (size_t)blockIdx.x * blockDim.x + threadIdx.x;
  if (i < n) out[i] = (_Float16)in[i];
}

// ---------------------------------------------------------------------------
// TDM: async-copy one 4KB B-stage (2048 f16, contiguous) global -> LDS.
// D# per CDNA5 ISA: 1-row tile, data_size=2B,
// tile_dim0 = tensor_dim0 = tensor_dim0_stride = 2048.
// ---------------------------------------------------------------------------
#if HAVE_TDM
__device__ __forceinline__ void tdm_load_4kb(const _Float16* gsrc, _Float16* ldst) {
  const unsigned lds = (unsigned)(uintptr_t)ldst;           // low 32b = LDS offset
  const unsigned long long ga = (unsigned long long)(uintptr_t)gsrc;
  u32x4 g0;
  g0[0] = 1u;                                               // count=1 (valid user D#)
  g0[1] = lds;                                              // lds_addr
  g0[2] = (unsigned)(ga & 0xFFFFFFFFu);                     // global_addr[31:0]
  g0[3] = (unsigned)((ga >> 32) & 0x01FFFFFFu) | (2u << 30); // addr[56:32] | type=2
  i32x8 g1;
  g1[0] = (int)(1u << 16);                                  // data_size=1 -> 2 bytes
  g1[1] = (int)((2048u & 0xFFFFu) << 16);                   // tensor_dim0 lo16 @bit48
  g1[2] = (int)((2048u >> 16) | (1u << 16));                // tdim0 hi | tensor_dim1=1
  g1[3] = (int)(2048u << 16);                               // tdim1 hi(0) | tile_dim0
  g1[4] = 1;                                                // tile_dim1=1, tile_dim2=0
  g1[5] = 2048;                                             // tensor_dim0_stride lo32
  g1[6] = 0;
  g1[7] = 0;
  i32x4 gz = {0, 0, 0, 0};
#if defined(__clang_major__) && (__clang_major__ >= 23)
  i32x8 gz8 = {0, 0, 0, 0, 0, 0, 0, 0};
  __builtin_amdgcn_tensor_load_to_lds(g0, g1, gz, gz, gz8, 0);
#else
  __builtin_amdgcn_tensor_load_to_lds(g0, g1, gz, gz, 0);
#endif
}
#endif

// ---------------------------------------------------------------------------
// WMMA GEMM: out[n,M] = A16[n,K] @ W + bias, W pre-packed (fragment order).
// Block = 8 waves; each wave owns one 16-row tile, all share the 4KB B-stage
// (4 col tiles) staged in LDS, double-buffered via the Tensor Data Mover.
// ---------------------------------------------------------------------------
__global__ __launch_bounds__(256)
void gemm_wmma_kernel(const _Float16* __restrict__ A, const _Float16* __restrict__ P,
                      const float* __restrict__ bias, void* __restrict__ outp,
                      int n, int K, int M, int out_half) {
  __shared__ _Float16 smem[2][2048];                        // 2 x 4KB stages
  const int tid   = threadIdx.x;
  const int lane  = tid & 31;
  const int wave  = tid >> 5;                               // 0..7
  const int l15   = lane & 15;
  const int hi    = lane >> 4;
  const int kbase = hi * 8;
  const int mt    = M >> 4;
  const int colG  = blockIdx.y;
  int rowT = blockIdx.x * 8 + wave;
  const int maxT = n / 16 - 1;
  if (rowT > maxT) rowT = maxT;                             // tail waves recompute last tile
  const int row = rowT * 16 + l15;
  const int T   = K >> 5;                                   // k-steps
  const _Float16* Pst = P + (size_t)colG * 4 * 512;         // + kt*mt*512 per stage
  const size_t stageStride = (size_t)mt * 512;

  // prologue: stage 0
#if HAVE_TDM
  if (wave == 0) tdm_load_4kb(Pst, &smem[0][0]);
#else
  {
    v8h t0 = *(const v8h*)(Pst + tid * 8);
    *(v8h*)(&smem[0][0] + tid * 8) = t0;
  }
#endif

  v8f acc[4] = {v8f{}, v8f{}, v8f{}, v8f{}};

  for (int kt = 0; kt < T; ++kt) {
    const int cu = kt & 1;
#if HAVE_TDM
    if (wave == 0) {
      if (kt + 1 < T) {
        tdm_load_4kb(Pst + (size_t)(kt + 1) * stageStride, &smem[cu ^ 1][0]);
        __builtin_amdgcn_s_wait_tensorcnt(1);   // older (stage kt) complete, next in flight
      } else {
        __builtin_amdgcn_s_wait_tensorcnt(0);
      }
    }
    __syncthreads();                            // stage cu visible to all waves
#else
    v8h nxt{};
    const bool have_next = (kt + 1 < T);
    if (have_next) nxt = *(const v8h*)(Pst + (size_t)(kt + 1) * stageStride + tid * 8);
    __syncthreads();
#endif

    // A fragment: two contiguous 16B loads per lane
    const _Float16* ar = A + (size_t)row * K + kt * 32;
    const v8h alo = *(const v8h*)(ar + kbase);
    const v8h ahi = *(const v8h*)(ar + 16 + kbase);
    const v16h a = __builtin_shufflevector(alo, ahi,
        0, 1, 2, 3, 4, 5, 6, 7, 8, 9, 10, 11, 12, 13, 14, 15);

#pragma unroll
    for (int t = 0; t < 4; ++t) {
      const _Float16* bp = &smem[cu][0] + t * 512 + lane * 16;   // ds_load_b128 x2
      const v8h blo = *(const v8h*)bp;
      const v8h bhi = *(const v8h*)(bp + 8);
      const v16h b = __builtin_shufflevector(blo, bhi,
          0, 1, 2, 3, 4, 5, 6, 7, 8, 9, 10, 11, 12, 13, 14, 15);
      acc[t] = __builtin_amdgcn_wmma_f32_16x16x32_f16(
          false, a, false, b, (short)0, acc[t], false, false);
    }
    __syncthreads();                            // done reading stage cu
#if !HAVE_TDM
    if (have_next) *(v8h*)(&smem[cu ^ 1][0] + tid * 8) = nxt;
#endif
  }

  // C/D layout: VGPR r -> row r (lanes 0-15) / r+8 (lanes 16-31); col = lane%16
  const int rbase = rowT * 16 + hi * 8;
  if (out_half) {
    _Float16* o = (_Float16*)outp;
#pragma unroll
    for (int t = 0; t < 4; ++t) {
      const int c = colG * 64 + t * 16 + l15;
      const float bv = bias[c];
#pragma unroll
      for (int r = 0; r < 8; ++r)
        o[(size_t)(rbase + r) * M + c] = (_Float16)(acc[t][r] + bv);
    }
  } else {
    float* o = (float*)outp;
#pragma unroll
    for (int t = 0; t < 4; ++t) {
      const int c = colG * 64 + t * 16 + l15;
      const float bv = bias[c];
#pragma unroll
      for (int r = 0; r < 8; ++r)
        o[(size_t)(rbase + r) * M + c] = acc[t][r] + bv;
    }
  }
}

// ---------------------------------------------------------------------------
// helpers / edge kernels
// ---------------------------------------------------------------------------
__device__ __forceinline__ int f2ord(float f) {
  int i = __float_as_int(f);
  return (i >= 0) ? i : (i ^ 0x7FFFFFFF);
}
__device__ __forceinline__ float ord2f(int i) {
  return __int_as_float((i >= 0) ? i : (i ^ 0x7FFFFFFF));
}

__global__ void fill_f32(float* __restrict__ p, float v, size_t n) {
  size_t i = (size_t)blockIdx.x * blockDim.x + threadIdx.x;
  if (i < n) p[i] = v;
}
__global__ void fill_i32(int* __restrict__ p, int v, size_t n) {
  size_t i = (size_t)blockIdx.x * blockDim.x + threadIdx.x;
  if (i < n) p[i] = v;
}

// per-(edge,head) attention score + segment max (ordered-int atomicMax)
__global__ void alpha_kernel(const float* __restrict__ q, const float* __restrict__ k,
                             const int* __restrict__ src, const int* __restrict__ dst,
                             float* __restrict__ alpha, int* __restrict__ amax,
                             int E, int Hh, int Cc, float scale) {
  size_t i = (size_t)blockIdx.x * blockDim.x + threadIdx.x;
  if (i >= (size_t)E * Hh) return;
  const int h = (int)(i % Hh);
  const int e = (int)(i / Hh);
  const int s = src[e], d = dst[e];
  const float4* q4 = (const float4*)(q + (size_t)d * Hh * Cc + (size_t)h * Cc);
  const float4* k4 = (const float4*)(k + (size_t)s * Hh * Cc + (size_t)h * Cc);
  float acc = 0.f;
#pragma unroll 4
  for (int c = 0; c < Cc / 4; ++c) {
    float4 a = q4[c], b = k4[c];
    acc += a.x * b.x + a.y * b.y + a.z * b.z + a.w * b.w;
  }
  acc *= scale;
  alpha[i] = acc;
  atomicMax(&amax[(size_t)d * Hh + h], f2ord(acc));
}

// ex = exp(alpha - max); segment-sum into denom
__global__ void exp_kernel(float* __restrict__ alpha, const int* __restrict__ amax,
                           float* __restrict__ denom, const int* __restrict__ dst,
                           int E, int Hh) {
  size_t i = (size_t)blockIdx.x * blockDim.x + threadIdx.x;
  if (i >= (size_t)E * Hh) return;
  const int h = (int)(i % Hh);
  const int e = (int)(i / Hh);
  const size_t dh = (size_t)dst[e] * Hh + h;
  const float ex = __expf(alpha[i] - ord2f(amax[dh]));
  alpha[i] = ex;
  atomicAdd(&denom[dh], ex);
}

// agg[dst,h,c] += (ex/denom) * v[src,h,c]
__global__ void agg_kernel(const float* __restrict__ alpha, const float* __restrict__ denom,
                           const float* __restrict__ v, const int* __restrict__ src,
                           const int* __restrict__ dst, float* __restrict__ agg,
                           int E, int Hh, int Cc) {
  size_t i = (size_t)blockIdx.x * blockDim.x + threadIdx.x;
  if (i >= (size_t)E * Hh * Cc) return;
  const int    c  = (int)(i % Cc);
  const size_t eh = i / Cc;
  const int    h  = (int)(eh % Hh);
  const int    e  = (int)(eh / Hh);
  const int s = src[e], d = dst[e];
  const float w = alpha[eh] / denom[(size_t)d * Hh + h];
  atomicAdd(&agg[(size_t)d * Hh * Cc + (size_t)h * Cc + c],
            w * v[(size_t)s * Hh * Cc + (size_t)h * Cc + c]);
}

// hout16 = (relu?) (agg + skip), written as f16 (next layer consumes it as GEMM A)
__global__ void combine_kernel(const float* __restrict__ agg, const float* __restrict__ skip,
                               _Float16* __restrict__ out, size_t n, int do_relu) {
  size_t i = (size_t)blockIdx.x * blockDim.x + threadIdx.x;
  if (i >= n) return;
  float v = agg[i] + skip[i];
  out[i] = (_Float16)(do_relu ? fmaxf(v, 0.f) : v);
}

// column mean over nodes: g[c] = mean(h[:, c])  (h stored f16)
__global__ void colmean_kernel(const _Float16* __restrict__ h, float* __restrict__ g,
                               int n, int M) {
  __shared__ float sdata[256];
  const int c = blockIdx.x;
  float acc = 0.f;
  for (int r = threadIdx.x; r < n; r += blockDim.x) acc += (float)h[(size_t)r * M + c];
  sdata[threadIdx.x] = acc;
  __syncthreads();
  for (int s = 128; s > 0; s >>= 1) {
    if ((int)threadIdx.x < s) sdata[threadIdx.x] += sdata[threadIdx.x + s];
    __syncthreads();
  }
  if (threadIdx.x == 0) g[c] = sdata[0] / (float)n;
}

// logits = relu(g @ Wc1 + bc1) @ Wc2 + bc2   (tiny: one block)
__global__ void mlp_kernel(const float* __restrict__ g,
                           const float* __restrict__ Wc1, const float* __restrict__ bc1,
                           const float* __restrict__ Wc2, const float* __restrict__ bc2,
                           float* __restrict__ out) {
  __shared__ float t1[32];
  const int j = threadIdx.x;
  if (j < 32) {
    float acc = bc1[j];
    for (int d = 0; d < 64; ++d) acc += g[d] * Wc1[d * 32 + j];
    t1[j] = fmaxf(acc, 0.f);
  }
  __syncthreads();
  if (j < 2) {
    float acc = bc2[j];
    for (int d = 0; d < 32; ++d) acc += t1[d] * Wc2[d * 2 + j];
    out[j] = acc;
  }
}

// ---------------------------------------------------------------------------
static inline unsigned nblk(size_t n) { return (unsigned)((n + 255) / 256); }

static void pack_w(const float* W, _Float16* P, int K, int M, hipStream_t stream) {
  pack_w_kernel<<<nblk((size_t)K * M), 256, 0, stream>>>(W, P, K, M);
}

static void gemm(const _Float16* A, const _Float16* P, const float* bias, void* out,
                 int K, int M, int out_half, hipStream_t stream) {
  dim3 gg((N_NODES / 16 + 7) / 8, M / 64);
  gemm_wmma_kernel<<<gg, 256, 0, stream>>>(A, P, bias, out, N_NODES, K, M, out_half);
}

static void run_conv(const _Float16* hin, _Float16* hout,
                     const _Float16* Pq, const float* bq, const _Float16* Pk, const float* bk,
                     const _Float16* Pv, const float* bv, const _Float16* Ps, const float* bs,
                     int Kin, int Hh, int Cc, bool relu,
                     float* qb, float* kb, float* vb, float* sb,
                     float* alpha, int* amax, float* denom,
                     const int* src, const int* dst, hipStream_t stream) {
  const int n = N_NODES, E = N_EDGES;
  const int Ml = Hh * Cc;
  gemm(hin, Pq, bq, qb, Kin, Ml, 0, stream);
  gemm(hin, Pk, bk, kb, Kin, Ml, 0, stream);
  gemm(hin, Pv, bv, vb, Kin, Ml, 0, stream);
  gemm(hin, Ps, bs, sb, Kin, Ml, 0, stream);

  const size_t nh = (size_t)n * Hh;
  fill_i32<<<nblk(nh), 256, 0, stream>>>(amax, (int)0x807FFFFFu, nh);  // ordered(-inf)
  fill_f32<<<nblk(nh), 256, 0, stream>>>(denom, 0.f, nh);

  const size_t eh = (size_t)E * Hh;
  const float scale = rsqrtf((float)Cc);
  alpha_kernel<<<nblk(eh), 256, 0, stream>>>(qb, kb, src, dst, alpha, amax, E, Hh, Cc, scale);
  exp_kernel<<<nblk(eh), 256, 0, stream>>>(alpha, amax, denom, dst, E, Hh);

  const size_t nm = (size_t)n * Ml;
  fill_f32<<<nblk(nm), 256, 0, stream>>>(qb, 0.f, nm);                 // agg reuses q buffer
  const size_t ehc = eh * (size_t)Cc;
  agg_kernel<<<nblk(ehc), 256, 0, stream>>>(alpha, denom, vb, src, dst, qb, E, Hh, Cc);
  combine_kernel<<<nblk(nm), 256, 0, stream>>>(qb, sb, hout, nm, relu ? 1 : 0);
}

extern "C" void kernel_launch(void* const* d_in, const int* in_sizes, int n_in,
                              void* d_out, int out_size, void* d_ws, size_t ws_size,
                              hipStream_t stream) {
  (void)in_sizes; (void)n_in; (void)out_size; (void)ws_size;
  const float* x   = (const float*)d_in[0];
  const int*   ei  = (const int*)d_in[1];
  const int*   src = ei;
  const int*   dst = ei + N_EDGES;
  const float* Wp  = (const float*)d_in[2];
  const float* bp  = (const float*)d_in[3];
  const float* W0[4] = {(const float*)d_in[4], (const float*)d_in[5],
                        (const float*)d_in[6], (const float*)d_in[7]};
  const float* b0[4] = {(const float*)d_in[8], (const float*)d_in[9],
                        (const float*)d_in[10], (const float*)d_in[11]};
  const float* Wl[4] = {(const float*)d_in[12], (const float*)d_in[13],
                        (const float*)d_in[14], (const float*)d_in[15]};
  const float* bl[4] = {(const float*)d_in[16], (const float*)d_in[17],
                        (const float*)d_in[18], (const float*)d_in[19]};
  const float* Wf[4] = {(const float*)d_in[20], (const float*)d_in[21],
                        (const float*)d_in[22], (const float*)d_in[23]};
  const float* bf[4] = {(const float*)d_in[24], (const float*)d_in[25],
                        (const float*)d_in[26], (const float*)d_in[27]};
  const float* Wc1 = (const float*)d_in[28];
  const float* bc1 = (const float*)d_in[29];
  const float* Wc2 = (const float*)d_in[30];
  const float* bc2 = (const float*)d_in[31];

  // ---------------- workspace carve-up ----------------
  char* base = (char*)d_ws;
  size_t off = 0;
  auto alloc = [&](size_t bytes) {
    void* p = base + off;
    off += (bytes + 255) & ~(size_t)255;
    return p;
  };
  const size_t NF = (size_t)N_NODES * HCW;           // 5.12M elems
  float* qb    = (float*)alloc(NF * 4);
  float* kb    = (float*)alloc(NF * 4);
  float* vb    = (float*)alloc(NF * 4);
  float* sb    = (float*)alloc(NF * 4);
  float* alpha = (float*)alloc((size_t)N_EDGES * NHEADS * 4);
  float* denom = (float*)alloc((size_t)N_NODES * NHEADS * 4);
  int*   amax  = (int*)alloc((size_t)N_NODES * NHEADS * 4);
  float* gmean = (float*)alloc(64 * 4);
  _Float16* x16  = (_Float16*)alloc((size_t)N_NODES * DIM_IN * 2);
  _Float16* hA16 = (_Float16*)alloc(NF * 2);
  _Float16* hB16 = (_Float16*)alloc(NF * 2);
  _Float16* Pp   = (_Float16*)alloc((size_t)DIM_IN * DD * 2);
  _Float16* P0[4];
  for (int i = 0; i < 4; ++i) P0[i] = (_Float16*)alloc((size_t)DD * HCW * 2);
  _Float16* Pl[4][4];   // [layer][q,k,v,s]
  for (int l = 0; l < 4; ++l)
    for (int i = 0; i < 4; ++i) Pl[l][i] = (_Float16*)alloc((size_t)HCW * HCW * 2);
  _Float16* Pf[4];
  for (int i = 0; i < 4; ++i) Pf[i] = (_Float16*)alloc((size_t)HCW * DD * 2);

  // ---------------- weight packing + input conversion ----------------
  cvt_f16_kernel<<<nblk((size_t)N_NODES * DIM_IN), 256, 0, stream>>>(
      x, x16, (size_t)N_NODES * DIM_IN);
  pack_w(Wp, Pp, DIM_IN, DD, stream);
  for (int i = 0; i < 4; ++i) pack_w(W0[i], P0[i], DD, HCW, stream);
  for (int l = 0; l < 4; ++l)
    for (int i = 0; i < 4; ++i)
      pack_w(Wl[i] + (size_t)l * HCW * HCW, Pl[l][i], HCW, HCW, stream);
  for (int i = 0; i < 4; ++i) pack_w(Wf[i], Pf[i], HCW, DD, stream);

  // ---------------- forward pass ----------------
  // input projection: hA16[N,64] = x @ Wp + bp
  gemm(x16, Pp, bp, hA16, DIM_IN, DD, 1, stream);

  // layer 0: [N,64] -> [N,512], relu
  run_conv(hA16, hB16, P0[0], b0[0], P0[1], b0[1], P0[2], b0[2], P0[3], b0[3],
           DD, NHEADS, DD, true, qb, kb, vb, sb, alpha, amax, denom, src, dst, stream);

  // layers 1..4: [N,512] -> [N,512], relu (ping-pong)
  _Float16* cur = hB16;
  _Float16* oth = hA16;
  for (int l = 0; l < 4; ++l) {
    run_conv(cur, oth, Pl[l][0], bl[0] + (size_t)l * HCW, Pl[l][1], bl[1] + (size_t)l * HCW,
             Pl[l][2], bl[2] + (size_t)l * HCW, Pl[l][3], bl[3] + (size_t)l * HCW,
             HCW, NHEADS, DD, true, qb, kb, vb, sb, alpha, amax, denom, src, dst, stream);
    _Float16* t = cur; cur = oth; oth = t;
  }

  // final conv: [N,512] -> [N,64], heads=1, no relu
  run_conv(cur, oth, Pf[0], bf[0], Pf[1], bf[1], Pf[2], bf[2], Pf[3], bf[3],
           HCW, 1, DD, false, qb, kb, vb, sb, alpha, amax, denom, src, dst, stream);

  // graph mean pool + classifier MLP
  colmean_kernel<<<64, 256, 0, stream>>>(oth, gmean, N_NODES, DD);
  mlp_kernel<<<1, 64, 0, stream>>>(gmean, Wc1, bc1, Wc2, bc2, (float*)d_out);
}
